// NearestNeighbourModule_72378788872239
// MI455X (gfx1250) — compile-verified, matrix operand order checked by reading, not run on hardware
//
#include <hip/hip_runtime.h>
#include <hip/hip_bf16.h>

typedef __attribute__((ext_vector_type(16))) _Float16 v16h;
typedef __attribute__((ext_vector_type(8)))  float    v8f;
typedef __attribute__((ext_vector_type(4)))  float    v4f;

// ---------------------------------------------------------------------------
// Kernel 1: row squared norms (sq[i] = sum_k x[i][k]^2), d = 64 fixed.
// ---------------------------------------------------------------------------
__global__ __launch_bounds__(256) void nn72378_sqnorm(const float* __restrict__ x,
                                                      float* __restrict__ sq,
                                                      int rows) {
    int t = blockIdx.x * blockDim.x + threadIdx.x;
    if (t >= rows) return;
    const v4f* p = (const v4f*)(x + (size_t)t * 64);
    float s = 0.0f;
#pragma unroll
    for (int i = 0; i < 16; ++i) {
        v4f v = p[i];
        s += v[0] * v[0] + v[1] * v[1] + v[2] * v[2] + v[3] * v[3];
    }
    sq[t] = s;
}

// Load one 16x32 f16 WMMA fragment (A layout; B is the symmetric N-major
// layout so the same loader works) from a row-major f32 row pointer.
// Lane role: row/col index = lane&15, c0 = (lane>=16)?8:0.
// Half h: 0..7 -> K = c0+h ; 8..15 -> K = c0+16+(h-8).
__device__ __forceinline__ v16h nn72378_frag(const float* __restrict__ row, int c0) {
    const v4f* p = (const v4f*)row;
    v4f q0 = p[c0 >> 2];
    v4f q1 = p[(c0 >> 2) + 1];
    v4f q2 = p[(c0 + 16) >> 2];
    v4f q3 = p[((c0 + 16) >> 2) + 1];
    v16h f;
#pragma unroll
    for (int i = 0; i < 4; ++i) {
        f[i]      = (_Float16)q0[i];
        f[4 + i]  = (_Float16)q1[i];
        f[8 + i]  = (_Float16)q2[i];
        f[12 + i] = (_Float16)q3[i];
    }
    return f;
}

// ---------------------------------------------------------------------------
// Kernel 2: fused distance + softmax + contraction. One wave per 16-row tile,
// 8 waves per block (128 rows). Loops over all m in 16-column WMMA tiles.
// Scores are <= 0 so exp() never overflows: no online-max needed.
// ---------------------------------------------------------------------------
__global__ __launch_bounds__(256) void nn72378_main(const float* __restrict__ x0,
                                                    const float* __restrict__ x1,
                                                    const float* __restrict__ y,
                                                    const float* __restrict__ alpha_p,
                                                    const float* __restrict__ sq0,
                                                    const float* __restrict__ sq1,
                                                    float* __restrict__ out,
                                                    int n, int m) {
    const int lane = threadIdx.x & 31;
    const int wave = threadIdx.x >> 5;
    const int row0 = blockIdx.x * 128 + wave * 16;
    if (row0 >= n) return;
    const int half = lane >> 4;    // 0: lanes 0-15, 1: lanes 16-31
    const int l16  = lane & 15;
    const int c0   = half * 8;     // K base per A/B 16-bit fragment layout

    // Fold -alpha * log2(e) into one constant: exp(-a*dist) = exp2(k*dist).
    const float k_exp2 = -alpha_p[0] * 1.4426950408889634f;

    // A fragments: x0 row (row0 + l16), K 0..31 and 32..63.
    const float* a_row = x0 + (size_t)(row0 + l16) * 64;
    const v16h a_lo = nn72378_frag(a_row, c0);
    const v16h a_hi = nn72378_frag(a_row + 32, c0);

    // C/D layout: VGPR r, this lane holds element (M = r + 8*half, N = l16).
    float s0r[8];
#pragma unroll
    for (int r = 0; r < 8; ++r) s0r[r] = sq0[row0 + half * 8 + r];

    float lsum[8], acc[8];
#pragma unroll
    for (int r = 0; r < 8; ++r) { lsum[r] = 0.0f; acc[r] = 0.0f; }

    for (int j0 = 0; j0 < m; j0 += 16) {
        // B fragments: x1 row (j0 + l16) = column N of B (x1 tile transposed).
        const float* b_row = x1 + (size_t)(j0 + l16) * 64;
        const v16h b_lo = nn72378_frag(b_row, c0);
        const v16h b_hi = nn72378_frag(b_row + 32, c0);

        v8f c = {};
        c = __builtin_amdgcn_wmma_f32_16x16x32_f16(false, a_lo, false, b_lo,
                                                   (short)0, c, false, false);
        c = __builtin_amdgcn_wmma_f32_16x16x32_f16(false, a_hi, false, b_hi,
                                                   (short)0, c, false, false);

        const float s1v = sq1[j0 + l16];
        const float yv  = y[j0 + l16];
#pragma unroll
        for (int r = 0; r < 8; ++r) {
            float d2 = s0r[r] + s1v - 2.0f * c[r];
            d2 = fmaxf(d2, 0.0f);
            // raw v_sqrt_f32 (approx, ~1 ulp) — avoids IEEE fixup sequence
            float dist = __builtin_amdgcn_sqrtf(d2);
            // raw v_exp_f32; k_exp2 <= 0 so e in (0,1], no overflow possible
            float e = __builtin_amdgcn_exp2f(k_exp2 * dist);
            lsum[r] += e;
            acc[r]  += e * yv;
        }
    }

    // Reduce across the 16-lane N-group (xor masks stay within the half).
#pragma unroll
    for (int r = 0; r < 8; ++r) {
        float l = lsum[r], a = acc[r];
#pragma unroll
        for (int off = 1; off < 16; off <<= 1) {
            l += __shfl_xor(l, off, 32);
            a += __shfl_xor(a, off, 32);
        }
        lsum[r] = l; acc[r] = a;
    }
    if (l16 == 0) {
#pragma unroll
        for (int r = 0; r < 8; ++r)
            out[row0 + half * 8 + r] = acc[r] / lsum[r];
    }
}

extern "C" void kernel_launch(void* const* d_in, const int* in_sizes, int n_in,
                              void* d_out, int out_size, void* d_ws, size_t ws_size,
                              hipStream_t stream) {
    const float* x0    = (const float*)d_in[0];
    const float* x1    = (const float*)d_in[1];
    const float* y     = (const float*)d_in[2];
    const float* alpha = (const float*)d_in[3];
    float* out = (float*)d_out;

    const int d = 64;
    const int n = in_sizes[0] / d;
    const int m = in_sizes[1] / d;

    float* sq0 = (float*)d_ws;
    float* sq1 = sq0 + n;

    nn72378_sqnorm<<<(n + 255) / 256, 256, 0, stream>>>(x0, sq0, n);
    nn72378_sqnorm<<<(m + 255) / 256, 256, 0, stream>>>(x1, sq1, m);

    dim3 grid((n + 127) / 128);
    nn72378_main<<<grid, 256, 0, stream>>>(x0, x1, y, alpha, sq0, sq1, out, n, m);
}
